// SwinAttention_35081292874364
// MI455X (gfx1250) — compile-verified, matrix operand order checked by reading, not run on hardware
//
#include <hip/hip_runtime.h>
#include <hip/hip_bf16.h>

typedef __bf16 bf16;
typedef __attribute__((ext_vector_type(8)))  bf16  v8bf;
typedef __attribute__((ext_vector_type(16))) bf16  v16bf;
typedef __attribute__((ext_vector_type(8)))  float v8f;
typedef __attribute__((ext_vector_type(4)))  int   v4i;

#define NSEQ   8192
#define CDIM   1024
#define MTOT   32768          // B*N = 4*8192
#define SHIFTS 4

#if defined(__has_builtin)
#if __has_builtin(__builtin_amdgcn_global_load_async_to_lds_b128)
#define USE_ASYNC_LDS 1
#endif
#endif

#ifdef USE_ASYNC_LDS
typedef __attribute__((address_space(1))) v4i g_v4i;   // global int4
typedef __attribute__((address_space(3))) v4i l_v4i;   // LDS int4
// one lane copies 16B global -> LDS, no VGPR staging, tracked by ASYNCcnt
__device__ __forceinline__ void async_cp16(const bf16* g, bf16* l) {
  __builtin_amdgcn_global_load_async_to_lds_b128(
      (g_v4i*)(v4i*)g, (l_v4i*)(v4i*)l, 0, 0);
}
#endif

// ---------------- pack x (apply -SHIFT roll) to bf16 ----------------
__global__ __launch_bounds__(256) void k_pack_x(const float* __restrict__ x,
                                                bf16* __restrict__ xb) {
  int g  = blockIdx.x * 256 + threadIdx.x;   // MTOT * (CDIM/8) threads
  int m  = g >> 7;                            // dst (rolled) row
  int c0 = (g & 127) << 3;
  int b  = m >> 13;
  int n  = m & (NSEQ - 1);
  int sr = (b << 13) | ((n + SHIFTS) & (NSEQ - 1));
  const float* xp = x + (size_t)sr * CDIM + c0;
  v8bf o;
#pragma unroll
  for (int i = 0; i < 8; ++i) o[i] = (bf16)xp[i];
  *(v8bf*)(xb + (size_t)m * CDIM + c0) = o;
}

// ------------- pack weight transposed: w[K x Nc] -> wT[Nc x K] bf16 -------------
__global__ __launch_bounds__(256) void k_pack_wT(const float* __restrict__ w,
                                                 bf16* __restrict__ wT, int Nc) {
  int g = blockIdx.x * 256 + threadIdx.x;    // Nc * 1024 threads
  int n = g >> 10;
  int k = g & 1023;
  wT[g] = (bf16)w[(size_t)k * Nc + n];
}

// ---------------- WMMA GEMM: out[M x NOUT] = A[M x 1024] * BwT[NOUT x 1024] + bias ----------------
// Tile 128x128 per block, 8 waves (2x4), each wave 64x32 = 4x2 fragments of 16x16.
template <int NOUT, bool FINAL>
__global__ __launch_bounds__(256, 1) void k_wmma_gemm(const bf16* __restrict__ A,
                                                      const bf16* __restrict__ Bw,
                                                      const float* __restrict__ bias,
                                                      bf16* __restrict__ outb,
                                                      float* __restrict__ outf) {
  constexpr int K = 1024, BK = 64, KT = K / BK;   // 16 k-tiles
#ifdef USE_ASYNC_LDS
  __shared__ bf16 sA[2][128 * BK];
  __shared__ bf16 sB[2][128 * BK];
#else
  __shared__ bf16 sA[1][128 * BK];
  __shared__ bf16 sB[1][128 * BK];
#endif

  const int tid  = threadIdx.x;
  const int lane = tid & 31;
  const int wv   = tid >> 5;
  const int l16  = lane & 15;
  const int half = lane >> 4;
  const int wm   = (wv >> 2) * 64;   // wave m offset in tile
  const int wn   = (wv & 3) * 32;    // wave n offset in tile
  const int m0   = blockIdx.y * 128;
  const int n0   = blockIdx.x * 128;

  v8f acc[4][2];
#pragma unroll
  for (int j = 0; j < 2; ++j) {
    float bn = bias[n0 + wn + j * 16 + l16];
#pragma unroll
    for (int i = 0; i < 4; ++i)
#pragma unroll
      for (int r = 0; r < 8; ++r) acc[i][j][r] = bn;
  }

  // fragment compute from LDS buffer `buf`
  auto compute = [&](int buf) {
#pragma unroll
    for (int ks = 0; ks < 2; ++ks) {          // two 16x16x32 steps per BK=64
      const int klo = ks * 32 + half * 8;
      const int khi = ks * 32 + 16 + half * 8;
      v16bf af[4], bfr[2];
#pragma unroll
      for (int i = 0; i < 4; ++i) {
        const bf16* p = &sA[buf][(wm + i * 16 + l16) * BK];
        v8bf lo = *(const v8bf*)(p + klo);
        v8bf hi = *(const v8bf*)(p + khi);
        af[i] = __builtin_shufflevector(lo, hi, 0,1,2,3,4,5,6,7,8,9,10,11,12,13,14,15);
      }
#pragma unroll
      for (int j = 0; j < 2; ++j) {
        const bf16* p = &sB[buf][(wn + j * 16 + l16) * BK];
        v8bf lo = *(const v8bf*)(p + klo);
        v8bf hi = *(const v8bf*)(p + khi);
        bfr[j] = __builtin_shufflevector(lo, hi, 0,1,2,3,4,5,6,7,8,9,10,11,12,13,14,15);
      }
#pragma unroll
      for (int i = 0; i < 4; ++i)
#pragma unroll
        for (int j = 0; j < 2; ++j)
          acc[i][j] = __builtin_amdgcn_wmma_f32_16x16x32_bf16(
              false, af[i], false, bfr[j], (short)0, acc[i][j], false, false);
    }
  };

#ifdef USE_ASYNC_LDS
  // async double-buffered pipeline: memory -> LDS with no VGPR staging
  auto issueTile = [&](int kt, int buf) {
#pragma unroll
    for (int i = 0; i < 4; ++i) {
      int c = tid + 256 * i;                  // 1024 chunks of 8 bf16 per array
      int row = c >> 3, kc = (c & 7) << 3;
      async_cp16(A  + (size_t)(m0 + row) * K + kt * BK + kc, &sA[buf][row * BK + kc]);
      async_cp16(Bw + (size_t)(n0 + row) * K + kt * BK + kc, &sB[buf][row * BK + kc]);
    }
  };

  issueTile(0, 0);
  asm volatile("s_wait_asynccnt 0x0" ::: "memory");
  __syncthreads();
  for (int kt = 0; kt < KT; ++kt) {
    const int buf = kt & 1;
    if (kt + 1 < KT) issueTile(kt + 1, buf ^ 1);   // overlap copy with compute
    compute(buf);
    if (kt + 1 < KT) asm volatile("s_wait_asynccnt 0x0" ::: "memory");
    __syncthreads();
  }
#else
  // fallback: single-buffer staging, no registers live across compute (no spills)
  for (int kt = 0; kt < KT; ++kt) {
    uint4 ra[4], rb[4];
#pragma unroll
    for (int i = 0; i < 4; ++i) {
      int c = tid + 256 * i;
      int row = c >> 3, kc = (c & 7) << 3;
      ra[i] = *(const uint4*)(A  + (size_t)(m0 + row) * K + kt * BK + kc);
      rb[i] = *(const uint4*)(Bw + (size_t)(n0 + row) * K + kt * BK + kc);
    }
    __syncthreads();                          // previous compute done
#pragma unroll
    for (int i = 0; i < 4; ++i) {
      int c = tid + 256 * i;
      int row = c >> 3, kc = (c & 7) << 3;
      *(uint4*)(&sA[0][row * BK + kc]) = ra[i];
      *(uint4*)(&sB[0][row * BK + kc]) = rb[i];
    }
    __syncthreads();
    compute(0);
  }
#endif

  // epilogue: C/D layout -> lane l16 = n, vgpr r -> m = half*8 + r
#pragma unroll
  for (int i = 0; i < 4; ++i) {
#pragma unroll
    for (int j = 0; j < 2; ++j) {
      const int n = n0 + wn + j * 16 + l16;
#pragma unroll
      for (int r = 0; r < 8; ++r) {
        const int m = m0 + wm + i * 16 + half * 8 + r;
        if constexpr (FINAL) {
          int b = m >> 13, nn = m & (NSEQ - 1);
          int dst = (b << 13) | ((nn + SHIFTS) & (NSEQ - 1));   // +SHIFT roll
          outf[(size_t)dst * NOUT + n] = acc[i][j][r];
        } else {
          outb[(size_t)m * NOUT + n] = (bf16)acc[i][j][r];
        }
      }
    }
  }
}

// ---------------- window attention: one block per 8-token window ----------------
// thread = (head h, query row r); scores 8x8, softmax, 8x64 output.
__global__ __launch_bounds__(128) void k_window_attn(const bf16* __restrict__ qkv,
                                                     const float* __restrict__ btab,
                                                     bf16* __restrict__ ab) {
  __shared__ bf16 sw[8 * 3072];                 // window q|k|v, 48 KB
  const int w   = blockIdx.x;                   // 0..4095
  const int tid = threadIdx.x;
  const bf16* src = qkv + (size_t)w * 8 * 3072;
#pragma unroll
  for (int i = 0; i < 24; ++i) {                // 3072 chunks of 8 bf16
    int c = tid + 128 * i;
    *(uint4*)(&sw[c * 8]) = *(const uint4*)(src + c * 8);
  }
  __syncthreads();

  const int h = tid >> 3, r = tid & 7;
  const bf16* qp = &sw[r * 3072 + h * 64];
  float s[8];
  float mx = -1e30f;
#pragma unroll
  for (int j = 0; j < 8; ++j) {
    const bf16* kp = &sw[j * 3072 + 1024 + h * 64];
    float a = 0.f;
#pragma unroll
    for (int d = 0; d < 64; ++d) a += (float)qp[d] * (float)kp[d];
    a = a * 0.125f + btab[(112 + r - j) * 16 + h];   // Swin bias: idx = 112 + i - j
    s[j] = a;
    mx = fmaxf(mx, a);
  }
  float sum = 0.f;
#pragma unroll
  for (int j = 0; j < 8; ++j) { s[j] = __expf(s[j] - mx); sum += s[j]; }
  const float inv = 1.f / sum;

  float o[64];
#pragma unroll
  for (int d = 0; d < 64; ++d) o[d] = 0.f;
#pragma unroll
  for (int j = 0; j < 8; ++j) {
    const float p = s[j] * inv;
    const bf16* vp = &sw[j * 3072 + 2048 + h * 64];
#pragma unroll
    for (int d = 0; d < 64; ++d) o[d] += p * (float)vp[d];
  }
  bf16* op = ab + (size_t)(w * 8 + r) * CDIM + h * 64;
#pragma unroll
  for (int d0 = 0; d0 < 64; d0 += 8) {
    v8bf v;
#pragma unroll
    for (int t = 0; t < 8; ++t) v[t] = (bf16)o[d0 + t];
    *(v8bf*)(op + d0) = v;
  }
}

extern "C" void kernel_launch(void* const* d_in, const int* in_sizes, int n_in,
                              void* d_out, int out_size, void* d_ws, size_t ws_size,
                              hipStream_t stream) {
  const float* x      = (const float*)d_in[0];
  const float* w_qkv  = (const float*)d_in[1];
  const float* b_qkv  = (const float*)d_in[2];
  const float* w_proj = (const float*)d_in[3];
  const float* b_proj = (const float*)d_in[4];
  const float* btab   = (const float*)d_in[5];
  float* out = (float*)d_out;

  char* p = (char*)d_ws;
  bf16* xb   = (bf16*)p; p += (size_t)MTOT * 1024 * 2;   //  64 MiB
  bf16* wqT  = (bf16*)p; p += (size_t)3072 * 1024 * 2;   //   6 MiB
  bf16* wpT  = (bf16*)p; p += (size_t)1024 * 1024 * 2;   //   2 MiB
  bf16* qkvb = (bf16*)p; p += (size_t)MTOT * 3072 * 2;   // 192 MiB
  bf16* ab   = (bf16*)p;                                 //  64 MiB

  k_pack_x <<<MTOT * (CDIM / 8) / 256, 256, 0, stream>>>(x, xb);
  k_pack_wT<<<3072 * 1024 / 256,       256, 0, stream>>>(w_qkv,  wqT, 3072);
  k_pack_wT<<<1024 * 1024 / 256,       256, 0, stream>>>(w_proj, wpT, 1024);

  k_wmma_gemm<3072, false><<<dim3(3072 / 128, MTOT / 128), 256, 0, stream>>>(
      xb, wqT, b_qkv, qkvb, nullptr);

  k_window_attn<<<MTOT / 8, 128, 0, stream>>>(qkvb, btab, ab);

  k_wmma_gemm<1024, true><<<dim3(1024 / 128, MTOT / 128), 256, 0, stream>>>(
      ab, wpT, b_proj, nullptr, out);
}
